// PlaceCells_41815801594299
// MI455X (gfx1250) — matrix operand order repaired and checked by reading, not compiled, not run on hardware
//
#include <hip/hip_runtime.h>

// ---- CDNA5 vector types ----
typedef __attribute__((ext_vector_type(16))) __bf16 v16bf;
typedef __attribute__((ext_vector_type(8)))  float  v8f;
typedef __attribute__((ext_vector_type(4)))  float  f32x4;
typedef __attribute__((ext_vector_type(4)))  __bf16 v4bf;
typedef __attribute__((ext_vector_type(4)))  unsigned int u32x4;
typedef __attribute__((ext_vector_type(8)))  int i32x8;
typedef __attribute__((ext_vector_type(4)))  int i32x4;

#define CELL_DIM  256
#define NUM_CELLS 2048
#define N_STATES  131072

#define CHUNK_CELLS  64                              // cells per TDM chunk
#define CHUNK_ELEMS  (CHUNK_CELLS * CELL_DIM)        // 16384 bf16
#define CHUNK_BYTES  (CHUNK_ELEMS * 2)               // 32 KB
#define NCHUNK       (NUM_CELLS / CHUNK_CELLS)       // 32
#define NT_PER_CHUNK (CHUNK_CELLS / 16)              // 4 N-tiles per chunk

// ---------------------------------------------------------------------------
// Kernel 1: convert placeCells f32 -> bf16 into workspace (1 MB, L2-resident).
// ---------------------------------------------------------------------------
__global__ void cvt_cells_bf16(const float* __restrict__ src,
                               __bf16* __restrict__ dst, int n) {
  int i = (blockIdx.x * blockDim.x + threadIdx.x) * 4;
  if (i < n) {
    f32x4 v = *(const f32x4*)(src + i);
    v4bf o;
    o[0] = (__bf16)v[0];
    o[1] = (__bf16)v[1];
    o[2] = (__bf16)v[2];
    o[3] = (__bf16)v[3];
    *(v4bf*)(dst + i) = o;
  }
}

// ---------------------------------------------------------------------------
// TDM: stage CHUNK_CELLS x CELL_DIM bf16 tile (32 KB) from global into LDS.
// D# per CDNA5 ISA (2-D tile, groups 2/3 zero).
// This toolchain exposes the 6-arg builtin form:
//   (u32x4 g0, i32x8 g1, i32x4 g2, i32x4 g3, i32x8 extra, i32 cpol)
// ---------------------------------------------------------------------------
__device__ __forceinline__ void tdm_load_cells_chunk(unsigned lds_off,
                                                     const __bf16* gptr) {
  unsigned long long ga = (unsigned long long)(uintptr_t)gptr;
  u32x4 g0;
  g0[0] = 1u;                                         // count=1 (valid user D#)
  g0[1] = lds_off;                                    // D#.lds_addr (bytes)
  g0[2] = (unsigned)ga;                               // global_addr[31:0]
  g0[3] = ((unsigned)(ga >> 32) & 0x01ffffffu)        // global_addr[56:32]
          | (2u << 30);                               // type=2 ("image")
  i32x8 g1;
  g1[0] = 1 << 16;                  // wg_mask=0 (no cluster); data_size=1 -> 2B
  g1[1] = (int)(CELL_DIM << 16);    // tensor_dim0[15:0]  @ bits[63:48] = 256
  g1[2] = (int)(NUM_CELLS << 16);   // dim0 hi=0; tensor_dim1[15:0] @ [95:80] = 2048
  g1[3] = (int)(CELL_DIM << 16);    // dim1 hi=0; tile_dim0 @ [127:112] = 256
  g1[4] = CHUNK_CELLS;              // tile_dim1 @ [143:128] = 64; tile_dim2=0
  g1[5] = CELL_DIM;                 // tensor_dim0_stride lo32 = 256 elems
  g1[6] = 0;                        // stride hi; tensor_dim1_stride (unused 2D)
  g1[7] = 0;
  i32x4 z4 = {0, 0, 0, 0};
  i32x8 z8 = {0, 0, 0, 0, 0, 0, 0, 0};
  __builtin_amdgcn_tensor_load_to_lds(g0, g1, z4, z4, z8, 0);
}

// ---------------------------------------------------------------------------
// Kernel 2: fused GEMM (x * cells^T) + row argmax.
// One wave32 owns a 16-row M tile (A register-resident, 64 VGPRs).
// All 8 waves share B chunks staged in LDS by the TDM, double-buffered.
// ---------------------------------------------------------------------------
__global__ __launch_bounds__(256)
void placecells_argmax_wmma(const float* __restrict__ x,
                            const __bf16* __restrict__ cells,
                            int* __restrict__ out) {
  extern __shared__ __bf16 smem[];   // 2 x 32 KB chunk buffers

  const int lane = threadIdx.x & 31;
  const int wave = threadIdx.x >> 5;
  const int half = lane >> 4;        // 0: lanes 0-15, 1: lanes 16-31
  const int l    = lane & 15;
  const int mtile = blockIdx.x * 8 + wave;
  const int rbase = mtile * 16;

  // ---- Load + convert A tile: 16 rows x 256 K, f32 -> bf16, contiguous k order (same as B).
  // Lanes 0-15 (row M=l):  elems 0..15 = K ks*32+0..15
  // Lanes 16-31 (row M=l): elems 0..15 = K ks*32+16..31
  v16bf a[8];
  const float* arow = x + (size_t)(rbase + l) * CELL_DIM;
#pragma unroll
  for (int ks = 0; ks < 8; ++ks) {
    const int k0 = ks * 32 + half * 16;
    const int k1 = ks * 32 + half * 16 + 8;
    f32x4 c0 = *(const f32x4*)(arow + k0);
    f32x4 c1 = *(const f32x4*)(arow + k0 + 4);
    f32x4 c2 = *(const f32x4*)(arow + k1);
    f32x4 c3 = *(const f32x4*)(arow + k1 + 4);
#pragma unroll
    for (int e = 0; e < 4; ++e) {
      a[ks][e]      = (__bf16)c0[e];
      a[ks][4 + e]  = (__bf16)c1[e];
      a[ks][8 + e]  = (__bf16)c2[e];
      a[ks][12 + e] = (__bf16)c3[e];
    }
  }

  float bestv[8];
  int   besti[8];
#pragma unroll
  for (int j = 0; j < 8; ++j) { bestv[j] = -__builtin_inff(); besti[j] = 0; }

  // Prologue: wave0 kicks off the TDM for chunk 0 into buffer 0.
  const unsigned lds_base = (unsigned)(uintptr_t)smem;   // LDS aperture: addr[31:0] = LDS offset
  if (wave == 0) tdm_load_cells_chunk(lds_base, cells);

  for (int ch = 0; ch < NCHUNK; ++ch) {
    // Wave0 waits its outstanding TDM (chunk ch); barrier publishes LDS to all
    // waves AND guarantees everyone is done reading the other buffer before
    // wave0 re-fills it below.
    if (wave == 0) __builtin_amdgcn_s_wait_tensorcnt(0);
    __syncthreads();
    if (wave == 0 && ch + 1 < NCHUNK)
      tdm_load_cells_chunk(lds_base + ((ch + 1) & 1) * CHUNK_BYTES,
                           cells + (size_t)(ch + 1) * CHUNK_ELEMS);

    const __bf16* lbuf = smem + (ch & 1) * CHUNK_ELEMS;  // row-major [cell][k]

#pragma unroll
    for (int t = 0; t < NT_PER_CHUNK; ++t) {
      // B 32x16 bf16 operand: lane needs lbuf[t*16 + l][ks*32 + half*16 .. +15]
      const __bf16* bp = lbuf + (t * 16 + l) * CELL_DIM + half * 16;
      v8f c = {};
#pragma unroll
      for (int ks = 0; ks < 8; ++ks) {
        v16bf b = *(const v16bf*)(bp + ks * 32);         // ds_load_b128 x2
        c = __builtin_amdgcn_wmma_f32_16x16x32_bf16(
            /*neg_a=*/false, a[ks], /*neg_b=*/false, b,
            /*c_mod=*/(short)0, c, /*reuse_a=*/false, /*reuse_b=*/false);
      }
      const int col = (ch * NT_PER_CHUNK + t) * 16 + l;
#pragma unroll
      for (int j = 0; j < 8; ++j) {
        if (c[j] > bestv[j]) { bestv[j] = c[j]; besti[j] = col; }
      }
    }
  }

  // ---- Cross-lane argmax reduction within each 16-lane half (xor masks < 16
  // never cross halves). Tie-break on lower index = jnp.argmax semantics.
#pragma unroll
  for (int j = 0; j < 8; ++j) {
    float v   = bestv[j];
    int   idx = besti[j];
#pragma unroll
    for (int m = 8; m >= 1; m >>= 1) {
      float ov = __shfl_xor(v, m, 32);
      int   oi = __shfl_xor(idx, m, 32);
      if (ov > v || (ov == v && oi < idx)) { v = ov; idx = oi; }
    }
    if (l == 0) out[rbase + half * 8 + j] = idx;  // lane0 -> rows 0..7, lane16 -> 8..15
  }
}

// ---------------------------------------------------------------------------
extern "C" void kernel_launch(void* const* d_in, const int* in_sizes, int n_in,
                              void* d_out, int out_size, void* d_ws, size_t ws_size,
                              hipStream_t stream) {
  const float* x     = (const float*)d_in[0];   // (131072, 256) f32
  const float* cells = (const float*)d_in[1];   // (2048, 256) f32
  __bf16* cellsBf = (__bf16*)d_ws;              // 1 MB scratch

  const int nc = NUM_CELLS * CELL_DIM;          // 524288
  cvt_cells_bf16<<<nc / 4 / 256, 256, 0, stream>>>(cells, cellsBf, nc);

  const int mtiles = N_STATES / 16;             // 8192 waves, 8 waves/block
  const size_t shmem = 2 * CHUNK_BYTES;         // 64 KB double-buffered B stage
  placecells_argmax_wmma<<<mtiles / 8, 256, shmem, stream>>>(x, cellsBf, (int*)d_out);
}
